// GCN_395136991497
// MI455X (gfx1250) — matrix operand order repaired and checked by reading, not compiled, not run on hardware
//
#include <hip/hip_runtime.h>
#include <hip/hip_bf16.h>

// ---------------- problem constants ----------------
constexpr int NN  = 50000;
constexpr int EE  = 800000;
constexpr int INF = 512;
constexpr int HIDF = 256;
constexpr int MIDF = 128;
constexpr int OUTF = 64;

typedef __attribute__((ext_vector_type(16))) __bf16 v16bf;
typedef __attribute__((ext_vector_type(8)))  float  v8f;

static inline int cdiv(long long a, long long b) { return (int)((a + b - 1) / b); }

// round-to-nearest-even f32 -> bf16 (as ushort bits)
__device__ __forceinline__ unsigned short f2bf(float f) {
    unsigned int u = __float_as_uint(f);
    u += 0x7FFFu + ((u >> 16) & 1u);
    return (unsigned short)(u >> 16);
}

// ---------------- simple elementwise kernels ----------------
__global__ void k_fill_f32(float* p, float v, int n) {
    int i = blockIdx.x * blockDim.x + threadIdx.x;
    if (i < n) p[i] = v;
}

__global__ void k_deg_count(const int* __restrict__ dst, float* __restrict__ deg, int e) {
    int i = blockIdx.x * blockDim.x + threadIdx.x;
    if (i < e) atomicAdd(&deg[dst[i]], 1.0f);
}

__global__ void k_rsqrt_inplace(float* p, int n) {
    int i = blockIdx.x * blockDim.x + threadIdx.x;
    if (i < n) p[i] = rsqrtf(p[i]);
}

__global__ void k_cvt_bf16(const float* __restrict__ in, unsigned short* __restrict__ out, int n) {
    int i = blockIdx.x * blockDim.x + threadIdx.x;
    if (i < n) out[i] = f2bf(in[i]);
}

__global__ void k_relu_cvt_bf16(const float* __restrict__ in, unsigned short* __restrict__ out, int n) {
    int i = blockIdx.x * blockDim.x + threadIdx.x;
    if (i < n) out[i] = f2bf(fmaxf(in[i], 0.0f));
}

// ---------------- weight packing into B-fragment-native layout ----------------
// Packed index p = ((kb*(F/16) + nt)*512 + lane*16 + idx), halves.
// Fragment semantics: b[idx] = W[kb*32 + (idx>>3)*16 + (lane>>4)*8 + (idx&7)][nt*16 + (lane&15)]
template<int K, int F>
__global__ void k_packW(const float* __restrict__ W, unsigned short* __restrict__ Wp) {
    int p = blockIdx.x * blockDim.x + threadIdx.x;
    if (p >= K * F) return;
    int idx  = p & 15;
    int lane = (p >> 4) & 31;
    int rest = p >> 9;
    int nt   = rest % (F / 16);
    int kb   = rest / (F / 16);
    int k = kb * 32 + ((idx >> 3) * 16) + ((lane >> 4) * 8) + (idx & 7);
    int n = nt * 16 + (lane & 15);
    Wp[p] = f2bf(W[(size_t)k * F + n]);
}

// ---------------- WMMA GEMM: out[N,F] = A[N,K](bf16) @ Wp(bf16, packed) (+bias) ----------------
// One wave computes a 16(M) x 64(N) strip: A-frag reused across 4 B-tiles.
union AFrag { v16bf v; uint4 q[2]; };

template<int K, int F>
__global__ __launch_bounds__(256) void k_gemm_wmma(const unsigned short* __restrict__ Abf,
                                                   const unsigned short* __restrict__ Wp,
                                                   const float* __restrict__ bias,
                                                   float* __restrict__ out) {
    constexpr int NQ = F / 64;                  // 64-wide n-strips
    const int w    = (blockIdx.x * blockDim.x + threadIdx.x) >> 5;
    const int lane = threadIdx.x & 31;
    const int tiles = (NN / 16) * NQ;
    if (w >= tiles) return;
    const int m0 = (w / NQ) * 16;
    const int n0 = (w % NQ) * 64;
    const int lr = lane & 15;                   // col-in-tile / row-in-tile (A)
    const int lh = lane >> 4;                   // lane half

    const unsigned short* arow = Abf + (size_t)(m0 + lr) * K;

    v8f acc[4] = {};
#pragma unroll 4
    for (int k0 = 0; k0 < K; k0 += 32) {
        AFrag a;
        // A layout: idx0..7 -> K = k0 + lh*8 .. +7 ; idx8..15 -> K = k0+16+lh*8 .. +7
        a.q[0] = *(const uint4*)(arow + k0 + lh * 8);
        a.q[1] = *(const uint4*)(arow + k0 + 16 + lh * 8);
        const unsigned short* wbase = Wp + ((size_t)(k0 >> 5) * (F / 16) + (n0 >> 4)) * 512 + lane * 16;
#pragma unroll
        for (int t = 0; t < 4; ++t) {
            v16bf b = *(const v16bf*)(const void*)(wbase + (size_t)t * 512);
            acc[t] = __builtin_amdgcn_wmma_f32_16x16x32_bf16(
                false, a.v, false, b, (short)0, acc[t], false, false);
        }
    }

#pragma unroll
    for (int t = 0; t < 4; ++t) {
        int col = n0 + t * 16 + lr;
        float bb = bias ? bias[col] : 0.0f;
#pragma unroll
        for (int v = 0; v < 8; ++v) {
            int row = m0 + lh * 8 + v;          // C/D layout: VGPR v, lane-half selects M / M+8
            out[(size_t)row * F + col] = acc[t][v] + bb;
        }
    }
}

// ---------------- GCN aggregation ----------------
// agg[i,f] = h[i,f] * dinv[i]^2 + b[f]   (self-loop term + bias)
template<int F>
__global__ void k_self_init(const float* __restrict__ h, const float* __restrict__ dinv,
                            const float* __restrict__ b, float* __restrict__ agg) {
    int i = blockIdx.x * blockDim.x + threadIdx.x;
    if (i >= NN * F) return;
    int row = i / F;
    int col = i - row * F;
    float di = dinv[row];
    agg[i] = h[i] * di * di + b[col];
}

// agg[dst] += h[src] * dinv[src]*dinv[dst], float4 per thread
template<int F>
__global__ void k_edge_scatter(const int* __restrict__ src, const int* __restrict__ dst,
                               const float* __restrict__ dinv, const float* __restrict__ h,
                               float* __restrict__ agg) {
    constexpr int C = F / 4;
    long long tid = (long long)blockIdx.x * blockDim.x + threadIdx.x;
    if (tid >= (long long)EE * C) return;
    int e = (int)(tid >> (F == 256 ? 6 : (F == 128 ? 5 : 4)));
    int c = (int)(tid & (C - 1));
    int s = src[e], d = dst[e];
    float norm = dinv[s] * dinv[d];
    const float4 hv = ((const float4*)(h + (size_t)s * F))[c];
    float* ap = agg + (size_t)d * F + c * 4;
    atomicAdd(ap + 0, hv.x * norm);
    atomicAdd(ap + 1, hv.y * norm);
    atomicAdd(ap + 2, hv.z * norm);
    atomicAdd(ap + 3, hv.w * norm);
}

// ---------------- in-place log_softmax over 64 cols, one wave per row ----------------
__global__ void k_log_softmax64(float* __restrict__ out, int n) {
    int w = (blockIdx.x * blockDim.x + threadIdx.x) >> 5;
    int lane = threadIdx.x & 31;
    if (w >= n) return;
    float* row = out + (size_t)w * 64;
    float a = row[lane], b = row[lane + 32];
    float m = fmaxf(a, b);
#pragma unroll
    for (int off = 16; off >= 1; off >>= 1) m = fmaxf(m, __shfl_xor(m, off, 32));
    float s = __expf(a - m) + __expf(b - m);
#pragma unroll
    for (int off = 16; off >= 1; off >>= 1) s += __shfl_xor(s, off, 32);
    float l = __logf(s);
    row[lane] = a - m - l;
    row[lane + 32] = b - m - l;
}

// ---------------- launch ----------------
extern "C" void kernel_launch(void* const* d_in, const int* in_sizes, int n_in,
                              void* d_out, int out_size, void* d_ws, size_t ws_size,
                              hipStream_t stream) {
    const float* x   = (const float*)d_in[0];
    const int*   ei  = (const int*)d_in[1];
    const float* W1  = (const float*)d_in[2];
    const float* b1  = (const float*)d_in[3];
    const float* W2  = (const float*)d_in[4];
    const float* b2  = (const float*)d_in[5];
    const float* Wc  = (const float*)d_in[6];
    const float* bc  = (const float*)d_in[7];
    float* out = (float*)d_out;
    const int* src = ei;
    const int* dst = ei + EE;

    // workspace carve-up (256B aligned)
    char* ws = (char*)d_ws;
    size_t off = 0;
    auto carve = [&](size_t bytes) { size_t o = off; off += (bytes + 255) & ~(size_t)255; return o; };
    float*          dinv = (float*)(ws + carve((size_t)NN * 4));
    unsigned short* Abf  = (unsigned short*)(ws + carve((size_t)NN * INF * 2));
    unsigned short* Wp   = (unsigned short*)(ws + carve((size_t)INF * HIDF * 2));
    float*          H    = (float*)(ws + carve((size_t)NN * HIDF * 4));
    float*          AGG  = (float*)(ws + carve((size_t)NN * HIDF * 4));

    const int B = 256;

    // degree -> dinv
    k_fill_f32<<<cdiv(NN, B), B, 0, stream>>>(dinv, 1.0f, NN);
    k_deg_count<<<cdiv(EE, B), B, 0, stream>>>(dst, dinv, EE);
    k_rsqrt_inplace<<<cdiv(NN, B), B, 0, stream>>>(dinv, NN);

    // ---- layer 1: h = x @ W1 ; agg = norm-propagate ; relu ----
    k_cvt_bf16<<<cdiv((long long)NN * INF, B), B, 0, stream>>>(x, Abf, NN * INF);
    k_packW<INF, HIDF><<<cdiv(INF * HIDF, B), B, 0, stream>>>(W1, Wp);
    {
        int waves = (NN / 16) * (HIDF / 64);
        k_gemm_wmma<INF, HIDF><<<cdiv((long long)waves * 32, B), B, 0, stream>>>(Abf, Wp, nullptr, H);
    }
    k_self_init<HIDF><<<cdiv((long long)NN * HIDF, B), B, 0, stream>>>(H, dinv, b1, AGG);
    k_edge_scatter<HIDF><<<cdiv((long long)EE * (HIDF / 4), B), B, 0, stream>>>(src, dst, dinv, H, AGG);
    k_relu_cvt_bf16<<<cdiv((long long)NN * HIDF, B), B, 0, stream>>>(AGG, Abf, NN * HIDF);

    // ---- layer 2: h = relu1 @ W2 ; propagate ; relu ----
    k_packW<HIDF, MIDF><<<cdiv(HIDF * MIDF, B), B, 0, stream>>>(W2, Wp);
    {
        int waves = (NN / 16) * (MIDF / 64);
        k_gemm_wmma<HIDF, MIDF><<<cdiv((long long)waves * 32, B), B, 0, stream>>>(Abf, Wp, nullptr, H);
    }
    k_self_init<MIDF><<<cdiv((long long)NN * MIDF, B), B, 0, stream>>>(H, dinv, b2, AGG);
    k_edge_scatter<MIDF><<<cdiv((long long)EE * (MIDF / 4), B), B, 0, stream>>>(src, dst, dinv, H, AGG);
    k_relu_cvt_bf16<<<cdiv((long long)NN * MIDF, B), B, 0, stream>>>(AGG, Abf, NN * MIDF);

    // ---- classifier: logits = relu2 @ Wc + bc -> log_softmax ----
    k_packW<MIDF, OUTF><<<cdiv(MIDF * OUTF, B), B, 0, stream>>>(Wc, Wp);
    {
        int waves = (NN / 16) * (OUTF / 64);
        k_gemm_wmma<MIDF, OUTF><<<cdiv((long long)waves * 32, B), B, 0, stream>>>(Abf, Wp, bc, out);
    }
    k_log_softmax64<<<cdiv((long long)NN * 32, B), B, 0, stream>>>(out, NN);
}
